// DFD_53309134078160
// MI455X (gfx1250) — matrix operand order repaired and checked by reading, not compiled, not run on hardware
//
#include <hip/hip_runtime.h>
#include <cstdint>
#include <cstddef>

#define NPIX   32768      // B*H*W
#define HWSZ   16384      // H*W
#define CCH    256        // channels
#define NCLS   80
#define CLSPAD 96         // 81 -> padded to 6*16
#define SMALLPAD 16
#define CAND   1000
#define MAXDET 100
#define BROW   40         // LDS row stride in halves (80B: conflict-free + 16B aligned)

typedef __attribute__((ext_vector_type(16))) _Float16 v16h;
typedef __attribute__((ext_vector_type(8)))  _Float16 v8h;
typedef __attribute__((ext_vector_type(8)))  float    v8f;
typedef unsigned long long u64;
typedef unsigned int       u32;

// ---------------- transpose: feat[B,C,HW] (f32) -> X0[(b*HW+p), C] (f16) ------
__global__ void k_transpose(const float* __restrict__ feat, _Float16* __restrict__ X0) {
  __shared__ float tile[32][33];
  const int p0 = blockIdx.x * 32, c0 = blockIdx.y * 32, b = blockIdx.z;
  const int tx = threadIdx.x, ty = threadIdx.y;
  tile[ty][tx] = feat[((size_t)(b * CCH + c0 + ty)) * HWSZ + p0 + tx];
  __syncthreads();
  X0[((size_t)(b * HWSZ + p0 + ty)) * CCH + c0 + tx] = (_Float16)tile[tx][ty];
}

// ---------------- weight / bias conversion with zero padding ----------------
__global__ void k_cvtw(const float* __restrict__ src, _Float16* __restrict__ dst,
                       int validRows, int padRows) {
  const int idx = blockIdx.x * blockDim.x + threadIdx.x;
  if (idx >= padRows * CCH) return;
  const int r = idx / CCH;
  dst[idx] = (r < validRows) ? (_Float16)src[idx] : (_Float16)0.f;
}

__global__ void k_cvtb(const float* __restrict__ src, float* __restrict__ dst,
                       int valid, int pad) {
  const int t = threadIdx.x;
  if (t < pad) dst[t] = (t < valid) ? src[t] : 0.f;
}

// ---------------- WMMA GEMM with LDS-staged weights ----------------
// Y[M,Npad] = X[M,256] * Wt[Npad,256]^T + bias (+res). f16 in, f32 accumulate.
// Block = 256 threads = 8 waves. All waves share ONE group of NT*16 columns;
// wave w handles M-tile (bm*8 + w). Per k-step the block stages the
// (NT*16 x 32) f16 weight panel into LDS once (8x less L2 weight traffic than
// per-wave global loads), then each wave reads B fragments via ds_load.
// A fragment (16-bit A 16x32): lane&15 = row; lanes<16 hold K 0..15,
// lanes>=16 hold K 16..31 -> one 32B global load.
// B fragment (16-bit B 32x16): lane&15 = column; lanes<16 hold K 0..15,
// lanes>=16 hold K 16..31 -> one 32B LDS read.
template <int NT>
__global__ void k_gemm(const _Float16* __restrict__ X, const _Float16* __restrict__ Wt,
                       const float* __restrict__ bias, const _Float16* __restrict__ res,
                       _Float16* __restrict__ Y, int M, int Npad, int Nvalid) {
  __shared__ _Float16 bs[NT * 16 * BROW];
  const int tid  = threadIdx.x;
  const int lane = tid & 31;
  const int ngroups = Npad / (NT * 16);
  const int bm = blockIdx.x / ngroups;          // block's M super-tile
  const int ng = blockIdx.x % ngroups;          // column group
  const int mtile = bm * 8 + (tid >> 5);        // one 16-row tile per wave
  const int m0 = mtile * 16, n0 = ng * NT * 16;
  const int row = m0 + (lane & 15);
  const int kbA = (lane >> 4) << 4;             // 0 or 16
  const int kbB = (lane >> 4) << 4;             // 0 or 16

  // staging assignment: thread -> (column, 16-half chunk)
  const int scol  = tid >> 1;                   // 0..127
  const int spart = tid & 1;                    // 0 or 1
  const bool sactive = (scol < NT * 16);
  const _Float16* srow = Wt + (size_t)(n0 + scol) * CCH + spart * 16;

  v8f acc[NT];
#pragma unroll
  for (int t = 0; t < NT; t++)
#pragma unroll
    for (int r = 0; r < 8; r++) acc[t][r] = 0.f;

  const _Float16* xrow = X + (size_t)row * CCH;
  for (int k0 = 0; k0 < CCH; k0 += 32) {
    __syncthreads();                            // previous panel fully consumed
    if (sactive)
      *(v16h*)&bs[scol * BROW + spart * 16] = *(const v16h*)(srow + k0);
    __syncthreads();                            // panel visible to all waves

    if (k0 + 32 < CCH)
      __builtin_prefetch(xrow + k0 + 32 + kbA, 0, 1);   // global_prefetch_b8
    v16h a = *(const v16h*)(xrow + k0 + kbA);
#pragma unroll
    for (int t = 0; t < NT; t++) {
      v16h bf = *(const v16h*)&bs[(t * 16 + (lane & 15)) * BROW + kbB];
      acc[t] = __builtin_amdgcn_wmma_f32_16x16x32_f16(
          false, a, false, bf, (short)0, acc[t], false, false);
    }
  }

  // C/D layout: VGPR r -> row m0+r (lanes<16) or m0+8+r (lanes>=16); col = lane&15
  const int ncol = lane & 15;
  const int mofs = (lane >> 4) << 3;
#pragma unroll
  for (int t = 0; t < NT; t++) {
    const int o = n0 + t * 16 + ncol;
    const float bv = bias[o];
#pragma unroll
    for (int r = 0; r < 8; r++) {
      const int m = m0 + mofs + r;
      float v = acc[t][r] + bv;
      if (res) v += (float)res[(size_t)m * CCH + o];
      if (o < Nvalid) Y[(size_t)m * Npad + o] = (_Float16)v;
    }
  }
}

// ---------------- decode: sigmoid/argmax/boxes + sort keys ----------------
__global__ void k_scorebox(const _Float16* __restrict__ cls, const _Float16* __restrict__ obj,
                           const _Float16* __restrict__ box, int* __restrict__ label,
                           float* __restrict__ boxes, u64* __restrict__ keys) {
  const int n = blockIdx.x * blockDim.x + threadIdx.x;
  if (n >= NPIX) return;
  const float ob = 1.f / (1.f + __expf(-(float)obj[(size_t)n * SMALLPAD]));
  const _Float16* cr = cls + (size_t)n * CLSPAD;
  float best = -1.f;
  int bl = 0;
  for (int c = 0; c < NCLS; c++) {
    const float s = ob / (1.f + __expf(-(float)cr[c]));
    if (s > best) { best = s; bl = c; }           // first max (argmax semantics)
  }
  const int hw = n & (HWSZ - 1);
  const int wx = hw & 127, hy = hw >> 7;
  const float px = (wx + 0.5f) * 8.f, py = (hy + 0.5f) * 8.f;
  const _Float16* br = box + (size_t)n * SMALLPAD;
  const float lft = __expf((float)br[0]) * 8.f, top = __expf((float)br[1]) * 8.f;
  const float rgt = __expf((float)br[2]) * 8.f, bot = __expf((float)br[3]) * 8.f;
  boxes[(size_t)n * 4 + 0] = px - lft;
  boxes[(size_t)n * 4 + 1] = py - top;
  boxes[(size_t)n * 4 + 2] = px + rgt;
  boxes[(size_t)n * 4 + 3] = py + bot;
  label[n] = bl;
  // score > 0 -> float bits order-preserving; ~hw makes ties prefer small index
  keys[n] = ((u64)__float_as_uint(best) << 32) | (u32)(~(u32)hw);
}

// ---------------- full descending bitonic sort, one block per batch ----------
__global__ void k_sort(u64* __restrict__ keys) {
  u64* kb = keys + ((size_t)blockIdx.x << 14);
  const int t = threadIdx.x;
  for (int k = 2; k <= HWSZ; k <<= 1) {
    for (int j = k >> 1; j > 0; j >>= 1) {
      for (int i = t; i < HWSZ; i += 1024) {
        const int l = i ^ j;
        if (l > i) {
          const u64 a = kb[i], c = kb[l];
          const bool desc = ((i & k) == 0);
          if (desc ? (a < c) : (a > c)) { kb[i] = c; kb[l] = a; }
        }
      }
      __syncthreads();
    }
  }
}

// ---------------- gather top-1000 candidates ----------------
__global__ void k_gather(const u64* __restrict__ keys, const int* __restrict__ label,
                         const float* __restrict__ boxes, float* __restrict__ cand_box,
                         float* __restrict__ cand_sc, int* __restrict__ cand_lab) {
  const int tid = blockIdx.x * blockDim.x + threadIdx.x;
  if (tid >= 2 * CAND) return;
  const int b = tid / CAND;
  const int j = tid % CAND;
  const u64 key = keys[(size_t)b * HWSZ + j];
  const u32 hw = ~((u32)key);
  const size_t n = (size_t)b * HWSZ + hw;
  cand_sc[tid] = __uint_as_float((u32)(key >> 32));
  cand_lab[tid] = label[n];
  for (int k = 0; k < 4; k++) cand_box[tid * 4 + k] = boxes[n * 4 + k];
}

// ---------------- greedy NMS (matches reference loop) ----------------
__global__ void k_nms(const float* __restrict__ cand_box, int* __restrict__ keepG) {
  __shared__ float x1[CAND], y1[CAND], x2[CAND], y2[CAND], ar[CAND];
  __shared__ int kp[CAND];
  const int b = blockIdx.x, t = threadIdx.x;
  for (int j = t; j < CAND; j += blockDim.x) {
    const float a0 = cand_box[(b * CAND + j) * 4 + 0];
    const float a1 = cand_box[(b * CAND + j) * 4 + 1];
    const float a2 = cand_box[(b * CAND + j) * 4 + 2];
    const float a3 = cand_box[(b * CAND + j) * 4 + 3];
    x1[j] = a0; y1[j] = a1; x2[j] = a2; y2[j] = a3;
    ar[j] = fmaxf(a2 - a0, 0.f) * fmaxf(a3 - a1, 0.f);
    kp[j] = 1;
  }
  __syncthreads();
  for (int i = 0; i < CAND - 1; i++) {
    if (kp[i] && t > i && t < CAND) {
      const float xx = fmaxf(x1[i], x1[t]);
      const float yy = fmaxf(y1[i], y1[t]);
      const float XX = fminf(x2[i], x2[t]);
      const float YY = fminf(y2[i], y2[t]);
      const float inter = fmaxf(XX - xx, 0.f) * fmaxf(YY - yy, 0.f);
      const float iou = inter / (ar[i] + ar[t] - inter + 1e-6f);
      if (iou > 0.65f) kp[t] = 0;
    }
    __syncthreads();
  }
  if (t < CAND) keepG[b * CAND + t] = kp[t];
}

// ---------------- finalize: stable partition, first 100 -> d_out ----------------
__global__ void k_finalize(const float* __restrict__ cand_box, const float* __restrict__ cand_sc,
                           const int* __restrict__ cand_lab, const int* __restrict__ keepG,
                           float* __restrict__ out) {
  __shared__ int scan[1024];
  const int b = blockIdx.x, t = threadIdx.x;
  const int kp = (t < CAND) ? keepG[b * CAND + t] : 0;
  scan[t] = kp;
  __syncthreads();
  for (int off = 1; off < 1024; off <<= 1) {
    const int v = (t >= off) ? scan[t - off] : 0;
    __syncthreads();
    scan[t] += v;
    __syncthreads();
  }
  const int total = scan[CAND - 1];
  if (t < CAND) {
    const int excl = scan[t] - kp;
    const int pos = kp ? excl : (total + (t - excl));
    if (pos < MAXDET) {
      float* det_box = out;
      float* det_sc  = out + 2 * MAXDET * 4;
      float* det_lab = out + 2 * MAXDET * 4 + 2 * MAXDET;
      for (int k = 0; k < 4; k++)
        det_box[(b * MAXDET + pos) * 4 + k] = cand_box[(b * CAND + t) * 4 + k];
      det_sc[b * MAXDET + pos]  = kp ? cand_sc[b * CAND + t] : 0.f;
      det_lab[b * MAXDET + pos] = (float)cand_lab[b * CAND + t];
    }
  }
}

// ---------------- host launcher ----------------
extern "C" void kernel_launch(void* const* d_in, const int* in_sizes, int n_in,
                              void* d_out, int out_size, void* d_ws, size_t ws_size,
                              hipStream_t stream) {
  (void)in_sizes; (void)n_in; (void)out_size; (void)ws_size;
  const float* feat      = (const float*)d_in[0];
  const float* cls_w_in  = (const float*)d_in[1];
  const float* cls_b_in  = (const float*)d_in[2];
  const float* cls_w_hid = (const float*)d_in[3];
  const float* cls_b_hid = (const float*)d_in[4];
  const float* cls_w_out = (const float*)d_in[5];
  const float* cls_b_out = (const float*)d_in[6];
  const float* obj_w_in  = (const float*)d_in[7];
  const float* obj_b_in  = (const float*)d_in[8];
  const float* obj_w_hid = (const float*)d_in[9];
  const float* obj_b_hid = (const float*)d_in[10];
  const float* obj_w_out = (const float*)d_in[11];
  const float* obj_b_out = (const float*)d_in[12];
  const float* box_w_in  = (const float*)d_in[13];
  const float* box_b_in  = (const float*)d_in[14];
  const float* box_w_hid = (const float*)d_in[15];
  const float* box_b_hid = (const float*)d_in[16];
  const float* box_w_out = (const float*)d_in[17];
  const float* box_b_out = (const float*)d_in[18];
  float* out = (float*)d_out;

  size_t off = 0;
  auto alloc = [&](size_t bytes) -> void* {
    void* p = (void*)((char*)d_ws + off);
    off += (bytes + 255) & ~(size_t)255;
    return p;
  };
  _Float16* X0   = (_Float16*)alloc((size_t)NPIX * CCH * 2);
  _Float16* bufA = (_Float16*)alloc((size_t)NPIX * CCH * 2);
  _Float16* bufB = (_Float16*)alloc((size_t)NPIX * CCH * 2);
  _Float16* wf_cls = (_Float16*)alloc((size_t)5 * CCH * CCH * 2);
  _Float16* wf_obj = (_Float16*)alloc((size_t)5 * CCH * CCH * 2);
  _Float16* wf_box = (_Float16*)alloc((size_t)5 * CCH * CCH * 2);
  _Float16* wo_cls = (_Float16*)alloc((size_t)CLSPAD * CCH * 2);
  _Float16* wo_obj = (_Float16*)alloc((size_t)SMALLPAD * CCH * 2);
  _Float16* wo_box = (_Float16*)alloc((size_t)SMALLPAD * CCH * 2);
  float* bo_cls = (float*)alloc(CLSPAD * 4);
  float* bo_obj = (float*)alloc(SMALLPAD * 4);
  float* bo_box = (float*)alloc(SMALLPAD * 4);
  _Float16* cls_out = (_Float16*)alloc((size_t)NPIX * CLSPAD * 2);
  _Float16* obj_out = (_Float16*)alloc((size_t)NPIX * SMALLPAD * 2);
  _Float16* box_out = (_Float16*)alloc((size_t)NPIX * SMALLPAD * 2);
  int*   label_buf = (int*)alloc((size_t)NPIX * 4);
  float* boxes_buf = (float*)alloc((size_t)NPIX * 4 * 4);
  u64*   keys      = (u64*)alloc((size_t)2 * HWSZ * 8);
  float* cand_box  = (float*)alloc(2 * CAND * 4 * 4);
  float* cand_sc   = (float*)alloc(2 * CAND * 4);
  int*   cand_lab  = (int*)alloc(2 * CAND * 4);
  int*   keepG     = (int*)alloc(2 * CAND * 4);

  // 1. channel-major f32 -> pixel-major f16
  k_transpose<<<dim3(HWSZ / 32, CCH / 32, 2), dim3(32, 32), 0, stream>>>(feat, X0);

  // 2. weights -> f16 (padded where needed)
  k_cvtw<<<CCH, 256, 0, stream>>>(cls_w_in, wf_cls, CCH, CCH);
  k_cvtw<<<4 * CCH, 256, 0, stream>>>(cls_w_hid, wf_cls + (size_t)CCH * CCH, 4 * CCH, 4 * CCH);
  k_cvtw<<<CLSPAD, 256, 0, stream>>>(cls_w_out, wo_cls, NCLS + 1, CLSPAD);
  k_cvtw<<<CCH, 256, 0, stream>>>(obj_w_in, wf_obj, CCH, CCH);
  k_cvtw<<<4 * CCH, 256, 0, stream>>>(obj_w_hid, wf_obj + (size_t)CCH * CCH, 4 * CCH, 4 * CCH);
  k_cvtw<<<SMALLPAD, 256, 0, stream>>>(obj_w_out, wo_obj, 2, SMALLPAD);
  k_cvtw<<<CCH, 256, 0, stream>>>(box_w_in, wf_box, CCH, CCH);
  k_cvtw<<<4 * CCH, 256, 0, stream>>>(box_w_hid, wf_box + (size_t)CCH * CCH, 4 * CCH, 4 * CCH);
  k_cvtw<<<SMALLPAD, 256, 0, stream>>>(box_w_out, wo_box, 4, SMALLPAD);
  k_cvtb<<<1, CLSPAD, 0, stream>>>(cls_b_out, bo_cls, NCLS + 1, CLSPAD);
  k_cvtb<<<1, SMALLPAD, 0, stream>>>(obj_b_out, bo_obj, 2, SMALLPAD);
  k_cvtb<<<1, SMALLPAD, 0, stream>>>(box_b_out, bo_box, 4, SMALLPAD);

  // 3. heads: in-conv, 4 residual convs, out-conv (all WMMA GEMMs)
  // grid = (M / 128 m-super-tiles) * ngroups
  auto run_head = [&](const _Float16* wf, const float* b_in, const float* b_hid,
                      const _Float16* wout, const float* bout, _Float16* hout,
                      int NpadOut, int NvalidOut) {
    const int mblocks = NPIX / 128;   // 256
    k_gemm<8><<<mblocks * 2, 256, 0, stream>>>(X0, wf, b_in, nullptr, bufA, NPIX, CCH, CCH);
    _Float16* cur = bufA;
    _Float16* nxt = bufB;
    for (int i = 0; i < 4; i++) {
      k_gemm<8><<<mblocks * 2, 256, 0, stream>>>(cur, wf + (size_t)(1 + i) * CCH * CCH,
                                                 b_hid + i * CCH, cur, nxt, NPIX, CCH, CCH);
      _Float16* tmp = cur; cur = nxt; nxt = tmp;
    }
    if (NpadOut == CLSPAD)
      k_gemm<6><<<mblocks, 256, 0, stream>>>(cur, wout, bout, nullptr, hout, NPIX, CLSPAD, NvalidOut);
    else
      k_gemm<1><<<mblocks, 256, 0, stream>>>(cur, wout, bout, nullptr, hout, NPIX, SMALLPAD, NvalidOut);
  };
  run_head(wf_cls, cls_b_in, cls_b_hid, wo_cls, bo_cls, cls_out, CLSPAD, NCLS + 1);
  run_head(wf_obj, obj_b_in, obj_b_hid, wo_obj, bo_obj, obj_out, SMALLPAD, 2);
  run_head(wf_box, box_b_in, box_b_hid, wo_box, bo_box, box_out, SMALLPAD, 4);

  // 4. decode scores/boxes + sort keys
  k_scorebox<<<NPIX / 256, 256, 0, stream>>>(cls_out, obj_out, box_out,
                                             label_buf, boxes_buf, keys);

  // 5. per-batch descending bitonic sort (single kernel, one block per batch)
  k_sort<<<2, 1024, 0, stream>>>(keys);

  // 6. top-1000 candidates, NMS, finalize top-100
  k_gather<<<(2 * CAND + 255) / 256, 256, 0, stream>>>(keys, label_buf, boxes_buf,
                                                       cand_box, cand_sc, cand_lab);
  k_nms<<<2, 1024, 0, stream>>>(cand_box, keepG);
  k_finalize<<<2, 1024, 0, stream>>>(cand_box, cand_sc, cand_lab, keepG, out);
}
